// DeepSIAUpBlock_72189810311252
// MI455X (gfx1250) — matrix operand order repaired and checked by reading, not compiled, not run on hardware
//
#include <hip/hip_runtime.h>

// ---------------------------------------------------------------- types
typedef __bf16 bfx8  __attribute__((ext_vector_type(8)));
typedef __bf16 bfx16 __attribute__((ext_vector_type(16)));
typedef float  fx8   __attribute__((ext_vector_type(8)));

#define B_   2
#define CIN  256
#define CC   384      // Cin + Cout
#define COUT 128
#define HS   64
#define WSZ  64
#define DSZ  16
#define HP   66       // padded H/W
#define NW   1024     // windows per batch (8*8*16)
#define T_   64       // tokens per window
#define NI   80       // instruction tokens
#define KV   144      // NI + T_
#define KVP  160      // padded to multiple of 32
#define HD   32       // head dim
#define NPOS (B_*DSZ*HS*WSZ)   // 131072 positions for stats

__device__ __forceinline__ fx8 zero8() {
  fx8 v;
#pragma unroll
  for (int i = 0; i < 8; ++i) v[i] = 0.0f;
  return v;
}

__device__ __forceinline__ fx8 wmma_bf16(bfx16 a, bfx16 b, fx8 c) {
  // v_wmma_f32_16x16x32_bf16 : D = A(16x32) x B(32x16) + C(16x16 f32)
  return __builtin_amdgcn_wmma_f32_16x16x32_bf16(false, a, false, b,
                                                 (short)0, c, false, false);
}

// build a 16-element bf16 fragment from two contiguous 8-element (16B) runs
__device__ __forceinline__ bfx16 frag2(const __bf16* p0, const __bf16* p1) {
  bfx8 lo = *(const bfx8*)p0;
  bfx8 hi = *(const bfx8*)p1;
  bfx16 r;
#pragma unroll
  for (int i = 0; i < 8; ++i) { r[i] = lo[i]; r[i + 8] = hi[i]; }
  return r;
}

// ---------------------------------------------------------------- utility
__global__ void k_zero(float* p, int n) {
  int i = blockIdx.x * blockDim.x + threadIdx.x;
  if (i < n) p[i] = 0.0f;
}

// ------------------------------------------------- upsample + concat + stats
// writes xa (B,D,66,66,384) bf16 interior; per-channel sum/sumsq via atomics
__global__ __launch_bounds__(256) void k_up_stats(
    const float* __restrict__ x, const float* __restrict__ xs,
    __bf16* __restrict__ xa, float* __restrict__ sums) {
  int c = blockIdx.y;
  int p = blockIdx.x * 256 + threadIdx.x;           // 0..131071
  int wo = p & 63, ho = (p >> 6) & 63, d = (p >> 12) & 15, b = p >> 16;
  float val;
  if (c < CIN) {
    int jh = ho >> 1, jw = wo >> 1;
    int h0, h1, w0, w1; float fh0, fh1, fw0, fw1;
    if (ho & 1) { h0 = jh; h1 = (jh + 1 > 31) ? 31 : jh + 1; fh0 = 0.75f; fh1 = 0.25f; }
    else        { h0 = (jh - 1 < 0) ? 0 : jh - 1; h1 = jh;   fh0 = 0.25f; fh1 = 0.75f; }
    if (wo & 1) { w0 = jw; w1 = (jw + 1 > 31) ? 31 : jw + 1; fw0 = 0.75f; fw1 = 0.25f; }
    else        { w0 = (jw - 1 < 0) ? 0 : jw - 1; w1 = jw;   fw0 = 0.25f; fw1 = 0.75f; }
    const float* xb = x + (((size_t)b * CIN + c) * 32 * 32) * 16 + d;
    float v00 = xb[(h0 * 32 + w0) * 16], v01 = xb[(h0 * 32 + w1) * 16];
    float v10 = xb[(h1 * 32 + w0) * 16], v11 = xb[(h1 * 32 + w1) * 16];
    val = fh0 * (fw0 * v00 + fw1 * v01) + fh1 * (fw0 * v10 + fw1 * v11);
  } else {
    val = xs[((((size_t)b * COUT + (c - CIN)) * HS + ho) * WSZ + wo) * DSZ + d];
  }
  xa[((((size_t)b * DSZ + d) * HP + (ho + 1)) * HP + (wo + 1)) * CC + c] = (__bf16)val;

  __shared__ float sb[256];
  sb[threadIdx.x] = val;
  __syncthreads();
  for (int s = 128; s > 0; s >>= 1) {
    if (threadIdx.x < s) sb[threadIdx.x] += sb[threadIdx.x + s];
    __syncthreads();
  }
  if (threadIdx.x == 0) atomicAdd(&sums[c], sb[0]);
  __syncthreads();
  sb[threadIdx.x] = val * val;
  __syncthreads();
  for (int s = 128; s > 0; s >>= 1) {
    if (threadIdx.x < s) sb[threadIdx.x] += sb[threadIdx.x + s];
    __syncthreads();
  }
  if (threadIdx.x == 0) atomicAdd(&sums[CC + c], sb[0]);
}

__global__ void k_finalize(const float* __restrict__ sums,
                           const float* __restrict__ gamma,
                           const float* __restrict__ beta,
                           float* __restrict__ sc, float* __restrict__ bi) {
  int c = threadIdx.x;
  if (c >= CC) return;
  float inv = 1.0f / (float)NPOS;
  float mu  = sums[c] * inv;
  float var = sums[CC + c] * inv - mu * mu;
  float rs  = rsqrtf(var + 1e-5f);
  float s   = gamma[c] * rs;
  sc[c] = s;
  bi[c] = beta[c] - mu * s;
}

// normalize + leaky-relu in place; zero the halo
__global__ __launch_bounds__(256) void k_normleaky(
    __bf16* __restrict__ xa, const float* __restrict__ sc,
    const float* __restrict__ bi) {
  size_t i = (size_t)blockIdx.x * 256 + threadIdx.x;
  const size_t total = (size_t)B_ * DSZ * HP * HP * CC;
  if (i >= total) return;
  int c = (int)(i % CC);
  size_t r = i / CC;
  int wp = (int)(r % HP); r /= HP;
  int hp = (int)(r % HP);
  if (hp >= 1 && hp <= 64 && wp >= 1 && wp <= 64) {
    float v = (float)xa[i] * sc[c] + bi[c];
    v = (v >= 0.0f) ? v : 0.01f * v;
    xa[i] = (__bf16)v;
  } else {
    xa[i] = (__bf16)0.0f;
  }
}

// pack conv weights tap-major k-contiguous, and qkv/proj to bf16
__global__ __launch_bounds__(256) void k_pack(
    const float* __restrict__ convw, const float* __restrict__ qkvw,
    const float* __restrict__ projw, __bf16* __restrict__ wpk,
    __bf16* __restrict__ qw, __bf16* __restrict__ pw) {
  int i = blockIdx.x * 256 + threadIdx.x;
  const int N0 = 9 * COUT * CC, N1 = 3 * COUT * COUT, N2 = COUT * COUT;
  if (i < N0) {
    int tap = i / (COUT * CC);
    int rem = i % (COUT * CC);
    int co = rem / CC, ci = rem % CC;
    wpk[i] = (__bf16)convw[((size_t)co * CC + ci) * 9 + tap];
  } else if (i < N0 + N1) {
    int j = i - N0; qw[j] = (__bf16)qkvw[j];
  } else if (i < N0 + N1 + N2) {
    int j = i - N0 - N1; pw[j] = (__bf16)projw[j];
  }
}

// instruction QKV (tiny): (B,80,384) bf16
__global__ __launch_bounds__(256) void k_qkvi(
    const float* __restrict__ xi, const float* __restrict__ qkvw,
    const float* __restrict__ qkvb, __bf16* __restrict__ qi) {
  int i = blockIdx.x * 256 + threadIdx.x;
  if (i >= B_ * NI * 3 * COUT) return;
  int n = i % (3 * COUT);
  int row = i / (3 * COUT);
  const float* xr = xi + (size_t)row * COUT;
  const float* wr = qkvw + (size_t)n * COUT;
  float acc = qkvb[n];
  for (int k = 0; k < COUT; ++k) acc += xr[k] * wr[k];
  qi[i] = (__bf16)acc;
}

// ------------------------------------------------- conv (implicit GEMM) + LN
// one block = one 8x8 window: M=64, N=128, K=9*384, WMMA bf16
__global__ __launch_bounds__(128) void k_conv_ln(
    const __bf16* __restrict__ xa, const __bf16* __restrict__ wpk,
    const float* __restrict__ convb, const float* __restrict__ lng,
    const float* __restrict__ lnb, float* __restrict__ res,
    __bf16* __restrict__ tn) {
  __shared__ float ybuf[T_][COUT + 4];
  int blk = blockIdx.x;
  int b = blk >> 10, widx = blk & 1023;
  int ih = widx >> 7, iw = (widx >> 4) & 7, d = widx & 15;
  int tid = threadIdx.x;
  int wave = tid >> 5, lane = tid & 31, lr = lane & 15, half = lane >> 4;
  int tok = wave * 16 + lr;
  int th = tok >> 3, tw = tok & 7;

  fx8 acc[8];
#pragma unroll
  for (int nt = 0; nt < 8; ++nt) acc[nt] = zero8();

#pragma unroll 1
  for (int tap = 0; tap < 9; ++tap) {
    int kh = tap / 3, kw = tap % 3;
    const __bf16* arow = xa +
        ((((size_t)b * DSZ + d) * HP + (ih * 8 + th + kh)) * HP +
         (iw * 8 + tw + kw)) * CC;
    if (tap < 8) {
      int kh2 = (tap + 1) / 3, kw2 = (tap + 1) % 3;
      __builtin_prefetch(xa + ((((size_t)b * DSZ + d) * HP + (ih * 8 + th + kh2)) * HP +
                               (iw * 8 + tw + kw2)) * CC, 0, 1);
    }
    const __bf16* wtap = wpk + (size_t)tap * COUT * CC;
#pragma unroll 4
    for (int kt = 0; kt < 12; ++kt) {
      bfx16 af = frag2(arow + kt * 32 + half * 16, arow + kt * 32 + half * 16 + 8);
#pragma unroll
      for (int nt = 0; nt < 8; ++nt) {
        const __bf16* brow = wtap + (size_t)(nt * 16 + lr) * CC + kt * 32 + half * 16;
        acc[nt] = wmma_bf16(af, frag2(brow, brow + 8), acc[nt]);
      }
    }
  }

#pragma unroll
  for (int nt = 0; nt < 8; ++nt) {
    int co = nt * 16 + lr;
    float cb = convb[co];
#pragma unroll
    for (int r = 0; r < 8; ++r)
      ybuf[wave * 16 + r + half * 8][co] = acc[nt][r] + cb;
  }
  __syncthreads();

  if (tid < T_) {
    int t = tid;
    float m = 0.0f;
#pragma unroll 8
    for (int c = 0; c < COUT; ++c) m += ybuf[t][c];
    m *= (1.0f / COUT);
    float v = 0.0f;
#pragma unroll 8
    for (int c = 0; c < COUT; ++c) { float dd = ybuf[t][c] - m; v += dd * dd; }
    float rs = rsqrtf(v * (1.0f / COUT) + 1e-5f);
    size_t base = ((size_t)blk * T_ + t) * COUT;
    for (int c = 0; c < COUT; ++c) {
      float y = ybuf[t][c];
      res[base + c] = y;
      tn[base + c] = (__bf16)((y - m) * rs * lng[c] + lnb[c]);
    }
  }
}

// ------------------------------------------------- windowed attention + proj
__global__ __launch_bounds__(128) void k_attn(
    const __bf16* __restrict__ tn, const float* __restrict__ res,
    const __bf16* __restrict__ qw, const float* __restrict__ qkvb,
    const __bf16* __restrict__ qi, const __bf16* __restrict__ pw,
    const float* __restrict__ projb, const float* __restrict__ biasc,
    const float* __restrict__ biasi, float* __restrict__ out) {
  __shared__ __align__(16) __bf16 qkvh[T_][96];      // q[0:32] k[32:64] v[64:96]
  __shared__ __align__(16) __bf16 vT[HD][KVP];
  __shared__ __align__(16) __bf16 Sbf[4][16][KVP];
  __shared__ __align__(16) __bf16 Obuf[T_][COUT];

  int blk = blockIdx.x;
  int b = blk >> 10, widx = blk & 1023;
  int ih = widx >> 7, iw = (widx >> 4) & 7, id = widx & 15;
  int tid = threadIdx.x;
  int wave = tid >> 5, lane = tid & 31, lr = lane & 15, half = lane >> 4;
  const __bf16* xwin = tn + (size_t)blk * T_ * COUT;
  const float scale = 0.17677669529663687f;           // 1/sqrt(32)

#pragma unroll 1
  for (int h = 0; h < 4; ++h) {
    __syncthreads();
    // ---- QKV GEMM for this head: wave owns 16 rows, 6 N-subtiles (q,k,v x2)
    {
      const __bf16* arow = xwin + (size_t)(wave * 16 + lr) * COUT;
#pragma unroll
      for (int s = 0; s < 6; ++s) {
        int sec = s >> 1, nn = s & 1;
        int gcol = sec * COUT + h * HD + nn * 16 + lr;
        int lcol = sec * 32 + nn * 16 + lr;
        fx8 accq = zero8();
        const __bf16* brow = qw + (size_t)gcol * COUT + half * 16;
#pragma unroll
        for (int kt = 0; kt < 4; ++kt) {
          bfx16 af = frag2(arow + kt * 32 + half * 16, arow + kt * 32 + half * 16 + 8);
          accq = wmma_bf16(af, frag2(brow + kt * 32, brow + kt * 32 + 8), accq);
        }
        float bv = qkvb[gcol];
#pragma unroll
        for (int r = 0; r < 8; ++r)
          qkvh[wave * 16 + r + half * 8][lcol] = (__bf16)(accq[r] + bv);
      }
    }
    __syncthreads();
    // ---- build V^T[dim][key] (instruction keys first, then content, pad 0)
    for (int idx = tid; idx < HD * KVP; idx += 128) {
      int dim = idx / KVP, key = idx % KVP;
      __bf16 v;
      if (key < NI)      v = qi[((size_t)b * NI + key) * 384 + 256 + h * HD + dim];
      else if (key < KV) v = qkvh[key - NI][64 + dim];
      else               v = (__bf16)0.0f;
      vT[dim][key] = v;
    }
    __syncthreads();
    // ---- S = q k^T (+scale +bias), register softmax, write bf16 P to LDS
    {
      fx8 s[9];
      const __bf16* qrow = &qkvh[wave * 16 + lr][0];
      bfx16 aq = frag2(qrow + half * 16, qrow + half * 16 + 8);
#pragma unroll
      for (int nt = 0; nt < 9; ++nt) {
        int key = nt * 16 + lr;
        const __bf16* bp = (key < NI)
            ? (qi + ((size_t)b * NI + key) * 384 + 128 + h * HD + half * 16)
            : (&qkvh[key - NI][32 + half * 16]);
        s[nt] = wmma_bf16(aq, frag2(bp, bp + 8), zero8());
      }
#pragma unroll
      for (int nt = 0; nt < 9; ++nt) {
        int key = nt * 16 + lr;
#pragma unroll
        for (int r = 0; r < 8; ++r) {
          int qt = wave * 16 + r + half * 8;
          float bv = (key < NI) ? biasi[((size_t)h * T_ + qt) * NI + key]
                                : biasc[((size_t)h * T_ + qt) * T_ + (key - NI)];
          s[nt][r] = s[nt][r] * scale + bv;
        }
      }
#pragma unroll
      for (int r = 0; r < 8; ++r) {
        float mx = -3.0e38f;
#pragma unroll
        for (int nt = 0; nt < 9; ++nt) mx = fmaxf(mx, s[nt][r]);
        for (int off = 1; off < 16; off <<= 1) mx = fmaxf(mx, __shfl_xor(mx, off, 32));
        float sum = 0.0f;
#pragma unroll
        for (int nt = 0; nt < 9; ++nt) {
          float e = __expf(s[nt][r] - mx);
          s[nt][r] = e; sum += e;
        }
        for (int off = 1; off < 16; off <<= 1) sum += __shfl_xor(sum, off, 32);
        float inv = 1.0f / sum;
        int row = r + half * 8;
#pragma unroll
        for (int nt = 0; nt < 9; ++nt)
          Sbf[wave][row][nt * 16 + lr] = (__bf16)(s[nt][r] * inv);
        Sbf[wave][row][KV + lr] = (__bf16)0.0f;   // pad keys 144..159
      }
    }
    // ---- O = P V  (K padded to 160), accumulate into Obuf columns of head h
    {
      fx8 o0 = zero8(), o1 = zero8();
#pragma unroll
      for (int kt = 0; kt < 5; ++kt) {
        const __bf16* sp = &Sbf[wave][lr][kt * 32 + half * 16];
        bfx16 as = frag2(sp, sp + 8);
        const __bf16* v0 = &vT[lr][kt * 32 + half * 16];
        const __bf16* v1 = &vT[16 + lr][kt * 32 + half * 16];
        o0 = wmma_bf16(as, frag2(v0, v0 + 8), o0);
        o1 = wmma_bf16(as, frag2(v1, v1 + 8), o1);
      }
#pragma unroll
      for (int r = 0; r < 8; ++r) {
        int t = wave * 16 + r + half * 8;
        Obuf[t][h * HD + lr]      = (__bf16)o0[r];
        Obuf[t][h * HD + 16 + lr] = (__bf16)o1[r];
      }
    }
  }
  __syncthreads();
  // ---- proj + bias + residual + scatter to (B,C,H,W,D)
  {
    const __bf16* orow = &Obuf[wave * 16 + lr][0];
    const float* resw = res + (size_t)blk * T_ * COUT;
#pragma unroll
    for (int nt = 0; nt < 8; ++nt) {
      int c = nt * 16 + lr;
      fx8 p = zero8();
      const __bf16* brow = pw + (size_t)c * COUT + half * 16;
#pragma unroll
      for (int kt = 0; kt < 4; ++kt) {
        bfx16 af = frag2(orow + kt * 32 + half * 16, orow + kt * 32 + half * 16 + 8);
        p = wmma_bf16(af, frag2(brow + kt * 32, brow + kt * 32 + 8), p);
      }
      float pb = projb[c];
#pragma unroll
      for (int r = 0; r < 8; ++r) {
        int t = wave * 16 + r + half * 8;
        float v = p[r] + pb + resw[(size_t)t * COUT + c];
        int hh = ih * 8 + (t >> 3), ww = iw * 8 + (t & 7);
        out[(((size_t)b * COUT + c) * HS + hh) * (WSZ * DSZ) + (size_t)ww * DSZ + id] = v;
      }
    }
  }
}

// ---------------------------------------------------------------- launcher
extern "C" void kernel_launch(void* const* d_in, const int* in_sizes, int n_in,
                              void* d_out, int out_size, void* d_ws, size_t ws_size,
                              hipStream_t stream) {
  (void)in_sizes; (void)n_in; (void)out_size; (void)ws_size;
  const float* x      = (const float*)d_in[0];
  const float* xs     = (const float*)d_in[1];
  const float* xi     = (const float*)d_in[2];
  const float* ngamma = (const float*)d_in[3];
  const float* nbeta  = (const float*)d_in[4];
  const float* convw  = (const float*)d_in[5];
  const float* convb  = (const float*)d_in[6];
  const float* lng    = (const float*)d_in[7];
  const float* lnb    = (const float*)d_in[8];
  const float* qkvw   = (const float*)d_in[9];
  const float* qkvb   = (const float*)d_in[10];
  const float* projw  = (const float*)d_in[11];
  const float* projb  = (const float*)d_in[12];
  const float* biasc  = (const float*)d_in[13];
  const float* biasi  = (const float*)d_in[14];
  float* out = (float*)d_out;

  char* ws = (char*)d_ws;
  size_t off = 0;
  auto alloc = [&](size_t bytes) -> void* {
    void* p = ws + off;
    off += (bytes + 255) & ~(size_t)255;
    return p;
  };
  float*  sums = (float*)alloc(2 * CC * sizeof(float));
  float*  scb  = (float*)alloc(2 * CC * sizeof(float));
  __bf16* xa   = (__bf16*)alloc((size_t)B_ * DSZ * HP * HP * CC * sizeof(__bf16));
  __bf16* wpk  = (__bf16*)alloc((size_t)9 * COUT * CC * sizeof(__bf16));
  __bf16* qw   = (__bf16*)alloc((size_t)3 * COUT * COUT * sizeof(__bf16));
  __bf16* pw   = (__bf16*)alloc((size_t)COUT * COUT * sizeof(__bf16));
  __bf16* qi   = (__bf16*)alloc((size_t)B_ * NI * 3 * COUT * sizeof(__bf16));
  float*  resb = (float*)alloc((size_t)B_ * NW * T_ * COUT * sizeof(float));
  __bf16* tnb  = (__bf16*)alloc((size_t)B_ * NW * T_ * COUT * sizeof(__bf16));

  k_zero<<<3, 256, 0, stream>>>(sums, 2 * CC);

  dim3 g1(512, CC);
  k_up_stats<<<g1, 256, 0, stream>>>(x, xs, xa, sums);

  k_finalize<<<1, 384, 0, stream>>>(sums, ngamma, nbeta, scb, scb + CC);

  size_t totp = (size_t)B_ * DSZ * HP * HP * CC;
  k_normleaky<<<(unsigned)((totp + 255) / 256), 256, 0, stream>>>(xa, scb, scb + CC);

  int packN = 9 * COUT * CC + 3 * COUT * COUT + COUT * COUT;
  k_pack<<<(packN + 255) / 256, 256, 0, stream>>>(convw, qkvw, projw, wpk, qw, pw);

  k_qkvi<<<(B_ * NI * 3 * COUT + 255) / 256, 256, 0, stream>>>(xi, qkvw, qkvb, qi);

  k_conv_ln<<<B_ * NW, 128, 0, stream>>>(xa, wpk, convb, lng, lnb, resb, tnb);

  k_attn<<<B_ * NW, 128, 0, stream>>>(tnb, resb, qw, qkvb, qi, pw, projb,
                                      biasc, biasi, out);
}
